// BitConv2dInfer_17600775979456
// MI455X (gfx1250) — matrix operand order repaired and checked by reading, not compiled, not run on hardware
//
#include <hip/hip_runtime.h>
#include <hip/hip_bf16.h>

typedef _Float16 v16h __attribute__((ext_vector_type(16)));
typedef _Float16 v8h  __attribute__((ext_vector_type(8)));
typedef _Float16 h2   __attribute__((ext_vector_type(2)));
typedef float    v8f  __attribute__((ext_vector_type(8)));

#define CN    32
#define CIN   256
#define COUT  256
#define HH    56
#define WW    56
#define HW    (HH*WW)          // 3136
#define PIX   (CN*HW)          // 100352
#define KTOT  (9*CIN)          // 2304

#define TILE_M 128             // cout per workgroup
#define TILE_N 128             // pixels per workgroup
#define TILE_K 32              // K per chunk (one tap x 32 cin)
#define NCHUNK (KTOT/TILE_K)   // 72
#define LDS_STRIDE 40          // f16 per row (32 data + 8 pad, keeps 16B align)

// ---------------------------------------------------------------------------
// Pre-pack ternary int32 weights -> f16, layout Wp[cout][tap*256 + cin]
// ---------------------------------------------------------------------------
__global__ __launch_bounds__(256)
void bitconv_prepack_w(const int* __restrict__ wq, _Float16* __restrict__ wp) {
    int idx = blockIdx.x * 256 + threadIdx.x;
    if (idx >= COUT * KTOT) return;
    int cout = idx / KTOT;
    int kk   = idx % KTOT;
    int tap  = kk / CIN;
    int cin  = kk % CIN;
    wp[idx] = (_Float16)(float)wq[(cout * CIN + cin) * 9 + tap];
}

// ---------------------------------------------------------------------------
// Implicit-GEMM conv: D[cout, pix] = W[cout, K] * A[K, pix]; f16 WMMA, f32 acc
// Weights staged memory->LDS with CDNA5 async-to-LDS DMA (ASYNCcnt).
// ---------------------------------------------------------------------------
__global__ __launch_bounds__(256)
void bitconv_gemm(const float* __restrict__ x,
                  const _Float16* __restrict__ wp,
                  const float* __restrict__ s,
                  const float* __restrict__ bias,
                  float* __restrict__ out) {
    __shared__ __align__(16) _Float16 sW[2][TILE_M * LDS_STRIDE]; // weights [cout][k]
    __shared__ __align__(16) _Float16 sA[2][TILE_N * LDS_STRIDE]; // acts    [pix][k]

    const int t        = threadIdx.x;
    const int pixBase  = blockIdx.x * TILE_N;
    const int coutBase = blockIdx.y * TILE_M;

    const int lane  = t & 31;
    const int wv    = t >> 5;
    const int waveM = wv >> 1;   // 0..3 : cout direction (32 cout each)
    const int waveN = wv & 1;    // 0..1 : pixel direction (64 pix each)
    const int lr    = lane & 15;
    const int hi    = lane >> 4; // 0 or 1

    // staging roles
    const int wrow  = t >> 1;    // 0..127 weight row (cout)
    const int wseg  = t & 1;     // which 32B half of the 64B k-row
    const int apix  = t & 127;   // 0..127 pixel row
    const int ahalf = t >> 7;    // 0..1 cin half

    // decompose this thread's staging pixel once
    const int p    = pixBase + apix;
    const int pn   = p / HW;
    const int prem = p % HW;
    const int py   = prem / WW;
    const int px   = prem % WW;

    v8f acc[2][4];
#pragma unroll
    for (int i = 0; i < 2; ++i)
#pragma unroll
        for (int j = 0; j < 4; ++j)
            acc[i][j] = (v8f)0.0f;

    auto load_chunk = [&](int c, int bf) {
        const int tap  = c >> 3;
        const int cinB = (c & 7) << 5;
        const int dy   = tap / 3 - 1;
        const int dx   = tap % 3 - 1;

        // ---- weights: async DMA global->LDS, no VGPR round trip ----
        {
            const _Float16* src =
                wp + (size_t)(coutBase + wrow) * KTOT + tap * CIN + cinB + wseg * 16;
            unsigned long long ga = (unsigned long long)(uintptr_t)src;
            // low 32 bits of a flat pointer to __shared__ == LDS byte address
            unsigned lda =
                (unsigned)(uintptr_t)&sW[bf][wrow * LDS_STRIDE + wseg * 16];
            // INST_OFFSET applies to both global and LDS sides (ISA 10.7/15.18)
            asm volatile("global_load_async_to_lds_b128 %0, %1, off"
                         :: "v"(lda), "v"(ga) : "memory");
            asm volatile("global_load_async_to_lds_b128 %0, %1, off offset:16"
                         :: "v"(lda), "v"(ga) : "memory");
        }

        // ---- activations: branchless gather, clamp, cvt f16, LDS store ----
        {
            const int yy = py + dy;
            const int xx = px + dx;
            const bool valid =
                ((unsigned)yy < (unsigned)HH) & ((unsigned)xx < (unsigned)WW);
            const int yyc = min(max(yy, 0), HH - 1);
            const int xxc = min(max(xx, 0), WW - 1);
            const float* xsrc =
                x + (((size_t)pn * CIN + cinB + ahalf * 16) * HH + yyc) * WW + xxc;
            _Float16* adst = &sA[bf][apix * LDS_STRIDE + ahalf * 16];
#pragma unroll
            for (int j = 0; j < 16; j += 2) {
                float f0 = xsrc[(size_t)j * HW];
                float f1 = xsrc[(size_t)(j + 1) * HW];
                f0 = valid ? f0 : 0.0f;          // v_cndmask, no exec divergence
                f1 = valid ? f1 : 0.0f;
                f0 = fminf(fmaxf(f0, -1.0f), 1.0f);
                f1 = fminf(fmaxf(f1, -1.0f), 1.0f);
                h2 v = { (_Float16)f0, (_Float16)f1 };
                *(h2*)(adst + j) = v;
            }
        }
    };

    load_chunk(0, 0);

    for (int c = 0; c < NCHUNK; ++c) {
        // async weight DMA must land in LDS before anyone reads past barrier
        asm volatile("s_wait_asynccnt 0x0" ::: "memory");
        __syncthreads();
        if (c + 1 < NCHUNK) load_chunk(c + 1, (c + 1) & 1);

        const int bf = c & 1;

        // A fragments (weights, 16x32 f16): lane<16 -> K0-7,K16-23 ; lane>=16 -> K8-15,K24-31
        v16h afrag[2];
#pragma unroll
        for (int tm = 0; tm < 2; ++tm) {
            const int row = (waveM * 32 + tm * 16 + lr) * LDS_STRIDE + hi * 8;
            v8h lo = *(const v8h*)&sW[bf][row];
            v8h up = *(const v8h*)&sW[bf][row + 16];
            afrag[tm] = __builtin_shufflevector(lo, up,
                0,1,2,3,4,5,6,7,8,9,10,11,12,13,14,15);
        }
        // B fragments (acts, 32x16 f16): lane<16 -> K0-7,K16-23 ; lane>=16 -> K8-15,K24-31
        v16h bfrag[4];
#pragma unroll
        for (int tn = 0; tn < 4; ++tn) {
            const int row = (waveN * 64 + tn * 16 + lr) * LDS_STRIDE + hi * 8;
            v8h lo = *(const v8h*)&sA[bf][row];
            v8h up = *(const v8h*)&sA[bf][row + 16];
            bfrag[tn] = __builtin_shufflevector(lo, up,
                0,1,2,3,4,5,6,7,8,9,10,11,12,13,14,15);
        }

#pragma unroll
        for (int tm = 0; tm < 2; ++tm)
#pragma unroll
            for (int tn = 0; tn < 4; ++tn)
                acc[tm][tn] = __builtin_amdgcn_wmma_f32_16x16x32_f16(
                    false, afrag[tm], false, bfrag[tn],
                    (short)0, acc[tm][tn], false, false);
    }

    // ---- epilogue: per-channel scale + bias, coalesced along pixels ----
#pragma unroll
    for (int tn = 0; tn < 4; ++tn) {
        const int pt   = pixBase + waveN * 64 + tn * 16 + lr;
        const int on   = pt / HW;
        const int orem = pt % HW;
        float* obase   = out + (size_t)on * COUT * HW + orem;
#pragma unroll
        for (int tm = 0; tm < 2; ++tm) {
            const int coutT = coutBase + waveM * 32 + tm * 16 + hi * 8;
#pragma unroll
            for (int r = 0; r < 8; ++r) {
                const int co = coutT + r;
                obase[(size_t)co * HW] = acc[tm][tn][r] * s[co] + bias[co];
            }
        }
    }
}

// ---------------------------------------------------------------------------
extern "C" void kernel_launch(void* const* d_in, const int* in_sizes, int n_in,
                              void* d_out, int out_size, void* d_ws, size_t ws_size,
                              hipStream_t stream) {
    const float* x    = (const float*)d_in[0];
    const int*   wq   = (const int*)d_in[1];
    const float* s    = (const float*)d_in[2];
    const float* bias = (const float*)d_in[3];
    float*       out  = (float*)d_out;
    _Float16*    wp   = (_Float16*)d_ws;   // 256*2304*2 = 1.18 MB of scratch

    bitconv_prepack_w<<<(COUT * KTOT + 255) / 256, 256, 0, stream>>>(wq, wp);

    dim3 grid(PIX / TILE_N, COUT / TILE_M);   // 784 x 2
    bitconv_gemm<<<grid, 256, 0, stream>>>(x, wp, s, bias, out);
}
